// WAVE_7224134992448
// MI455X (gfx1250) — matrix operand order repaired and checked by reading, not compiled, not run on hardware
//
#include <hip/hip_runtime.h>

#define D 256
#define NN 1000
#define NTHREADS 1024

typedef __attribute__((ext_vector_type(16))) _Float16 v16h;
typedef __attribute__((ext_vector_type(8)))  float    v8f;
typedef __attribute__((ext_vector_type(4)))  unsigned int u32x4;

union Frag16 { v16h h; u32x4 q[2]; };

// (toff, t, coff, c) for forward (rows 0-5) and backward (rows 6-11) sweeps
__constant__ int LVLS[12][4] = {
  {0,1,1,3},{1,3,4,9},{4,9,13,27},{13,27,40,81},{40,81,121,243},{121,243,364,636},
  {0,636,636,243},{636,243,879,81},{879,81,960,27},{960,27,987,9},{987,9,996,3},{996,3,999,1}
};

// ---------------- prep: copy h, build all f16 weight buffers (row-major, K contiguous)
__global__ void prep_kernel(const float* __restrict__ x,
                            const float* __restrict__ W_TA, const float* __restrict__ W_CA,
                            const float* __restrict__ W_TB, const float* __restrict__ W_CB,
                            const float* __restrict__ W_u,  const float* __restrict__ W_o,
                            float* __restrict__ hbuf,
                            _Float16* __restrict__ wta1, _Float16* __restrict__ wtb1,
                            _Float16* __restrict__ wc1h,     // [WCA1;WCB1] = GEMM weights AND S3
                            _Float16* __restrict__ S1h,      // [WTA2;WCA2;WTB2;WCB2;Wu2;Wo2] 1536x256
                            _Float16* __restrict__ S2h)      // [Wu1;Wo1] 512x256
{
  int gid = blockIdx.x * blockDim.x + threadIdx.x;
  int gsz = gridDim.x * blockDim.x;
  for (int i = gid; i < NN * D; i += gsz) hbuf[i] = x[i];  // x is (N,D) row-major == node-major h
  for (int i = gid; i < D * D; i += gsz) {
    int r = i >> 8, k = i & 255;
    wta1[i]            = (_Float16)W_TA[r*512 + k];
    wtb1[i]            = (_Float16)W_TB[r*512 + k];
    wc1h[i]            = (_Float16)W_CA[r*512 + k];        // WCA1
    wc1h[65536 + i]    = (_Float16)W_CB[r*512 + k];        // WCB1
    S1h[i]             = (_Float16)W_TA[r*512 + 256 + k];  // WTA2
    S1h[65536 + i]     = (_Float16)W_CA[r*512 + 256 + k];  // WCA2
    S1h[2*65536 + i]   = (_Float16)W_TB[r*512 + 256 + k];  // WTB2
    S1h[3*65536 + i]   = (_Float16)W_CB[r*512 + 256 + k];  // WCB2
    S1h[4*65536 + i]   = (_Float16)W_u [r*512 + 256 + k];  // Wu2
    S1h[5*65536 + i]   = (_Float16)W_o [r*512 + 256 + k];  // Wo2
    S2h[i]             = (_Float16)W_u [r*512 + k];        // Wu1
    S2h[65536 + i]     = (_Float16)W_o [r*512 + k];        // Wo1
  }
}

// ---------------- main: single persistent workgroup, full recurrence
__global__ __launch_bounds__(NTHREADS, 1)
void tree_main(float* __restrict__ hbuf, _Float16* __restrict__ act,
               const _Float16* __restrict__ wta1, const _Float16* __restrict__ wtb1,
               const _Float16* __restrict__ wc1h,
               const _Float16* __restrict__ S1h, const _Float16* __restrict__ S2h,
               const float* __restrict__ Wv,
               const float* __restrict__ bTA, const float* __restrict__ bCA,
               const float* __restrict__ bTB, const float* __restrict__ bCB,
               const float* __restrict__ bu,  const float* __restrict__ bo,
               float* __restrict__ QA, float* __restrict__ QB,
               float* __restrict__ PAb, float* __restrict__ PBb,
               float* __restrict__ out)
{
  __shared__ __align__(16) _Float16 s_h[256];    // f16 mirror of s
  __shared__ __align__(16) _Float16 m_h[256];    // f16 mirror of m
  __shared__ __align__(16) _Float16 new_h[256];  // f16 mirror of new
  __shared__ float v_lds[1536];
  __shared__ float m_lds[256];
  __shared__ float uo_lds[512];
  __shared__ float p1[1024], p2[1024], p3[1024];

  const int tid  = threadIdx.x;
  const int lane = tid & 31;
  const int wid  = tid >> 5;
  const _Float16* wca1 = wc1h;            // S3 == [WCA1;WCB1]
  const u32x4 zq = {0u, 0u, 0u, 0u};

  // W(16 rows) @ vec(256) via WMMA: vector in column 0 of B; returns D column-0 accum
  auto matvec_tile = [&](const _Float16* __restrict__ Wh, const _Float16* __restrict__ vh,
                         int tile) -> v8f {
    const int m   = tile * 16 + (lane & 15);
    const int kbA = (lane >> 4) * 8;
    const int kbB = (lane >> 4) * 8;
    const bool bl = (lane & 15) == 0;     // only lanes 0/16 carry the vector (N==0)
    v8f acc = {0.f,0.f,0.f,0.f,0.f,0.f,0.f,0.f};
    #pragma unroll
    for (int kc = 0; kc < 8; ++kc) {
      const int k0 = kc * 32;
      Frag16 fa, fb;
      const _Float16* ap = Wh + m * D + k0 + kbA;
      fa.q[0] = *(const u32x4*)(ap);
      fa.q[1] = *(const u32x4*)(ap + 16);
      if (bl) {
        const _Float16* bp = vh + k0 + kbB;
        fb.q[0] = *(const u32x4*)(bp);
        fb.q[1] = *(const u32x4*)(bp + 16);
      } else {
        fb.q[0] = zq; fb.q[1] = zq;
      }
      acc = __builtin_amdgcn_wmma_f32_16x16x32_f16(false, fa.h, false, fb.h,
                                                   (short)0, acc, false, false);
    }
    return acc;
  };

  auto biasf = [&](int row) -> float {
    const int r = row & 255, s = row >> 8;
    return (s==0) ? bTA[r] : (s==1) ? bCA[r] : (s==2) ? bTB[r]
         : (s==3) ? bCB[r] : (s==4) ? bu[r]  : bo[r];
  };

  for (int step = 0; step < 24; ++step) {            // 4 sweeps x 6 levels
    const int row  = ((step / 6) & 1) * 6 + (step % 6);
    const int toff = LVLS[row][0], t = LVLS[row][1];
    const int coff = LVLS[row][2], c = LVLS[row][3];
    const int tc = t + c;

    // ---- phase 1: refresh f16 activation mirror for T and C regions
    for (int i = tid; i < tc * D; i += NTHREADS) {
      int col = i >> 8, d = i & 255;
      int node = (col < t) ? (toff + col) : (coff + (col - t));
      act[node * D + d] = (_Float16)hbuf[node * D + d];
    }
    __syncthreads();

    // ---- phase 2: WMMA GEMMs  QA/QB = W?1 @ T,  PA/PB = WC?1 @ C   (f16 in, f32 out)
    {
      const int tTiles = (t + 15) >> 4, cTiles = (c + 15) >> 4;
      const int jobsQ = 2 * 16 * tTiles;
      const int jobsTot = jobsQ + 2 * 16 * cTiles;
      const int kbA = (lane >> 4) * 8;     // A frag: lanes 16-31 hold K+8 window
      const int kbB = (lane >> 4) * 8;     // B frag: lanes 16-31 hold K+8 window
      for (int job = wid; job < jobsTot; job += 32) {
        int jj = job;
        const int isP = (jj >= jobsQ);
        if (isP) jj -= jobsQ;
        const int sel = jj & 1; jj >>= 1;
        const int mt = jj & 15, ct = jj >> 4;
        const _Float16* wmat; float* dst; int nodebase, ncols;
        if (!isP) { wmat = sel ? wtb1 : wta1; dst = sel ? QB : QA;  nodebase = toff; ncols = t; }
        else      { wmat = sel ? (wc1h + 65536) : wca1; dst = sel ? PBb : PAb; nodebase = coff; ncols = c; }
        const int nloc = ct * 16 + (lane & 15);
        const int node = nodebase + ((nloc < ncols) ? nloc : 0);   // clamp: garbage cols not stored
        const int m    = mt * 16 + (lane & 15);
        v8f acc = {0.f,0.f,0.f,0.f,0.f,0.f,0.f,0.f};
        #pragma unroll
        for (int kc = 0; kc < 8; ++kc) {
          const int k0 = kc * 32;
          Frag16 fa, fb;
          const _Float16* ap = wmat + m * D + k0 + kbA;
          fa.q[0] = *(const u32x4*)(ap);
          fa.q[1] = *(const u32x4*)(ap + 16);     // K window +16
          const _Float16* bp = act + node * D + k0 + kbB;
          fb.q[0] = *(const u32x4*)(bp);
          fb.q[1] = *(const u32x4*)(bp + 16);
          acc = __builtin_amdgcn_wmma_f32_16x16x32_f16(false, fa.h, false, fb.h,
                                                       (short)0, acc, false, false);
        }
        if (nloc < ncols) {
          float* dp = dst + nloc * D + mt * 16 + (lane >> 4) * 8;  // col-major, 8 contiguous rows
          #pragma unroll
          for (int i = 0; i < 8; ++i) dp[i] = acc[i];
        }
      }
    }
    __syncthreads();

    // ---- sequential scan over child columns
    for (int k = 0; k < c; ++k) {
      // phase 3: s = C[:,k] (f16 mirror for WMMA matvec) + prefetch next column
      if (tid < 256) s_h[tid] = (_Float16)hbuf[(coff + k) * D + tid];
      if (tid < 8 && (k + 1) < c)
        __builtin_prefetch(hbuf + (coff + k + 1) * D + tid * 32, 0, 3);
      __syncthreads();

      // phase 4: v = S1 @ s (+ biases) on the matrix pipe: 96 tiles of 16 rows
      for (int tile = wid; tile < 96; tile += 32) {
        v8f acc = matvec_tile(S1h, s_h, tile);
        if ((lane & 15) == 0) {
          const int rbase = tile * 16 + (lane >> 4) * 8;
          #pragma unroll
          for (int i = 0; i < 8; ++i) {
            const int r = rbase + i;
            v_lds[r] = acc[i] + biasf(r);
          }
        }
      }
      __syncthreads();

      // phase 5: row softmax/softsign reductions -> m  (4 threads per row)
      {
        const int r = tid & 255, q = tid >> 8;
        const float tA = v_lds[r],       cA = v_lds[256 + r];
        const float tB = v_lds[512 + r], cB = v_lds[768 + r];
        float lm = -3.4e38f;
        for (int j = q; j < tc; j += 4) {
          float la = (j < t) ? QA[j*D + r] + tA : PAb[(j - t)*D + r] + cA;
          lm = fmaxf(lm, la);
        }
        p1[tid] = lm;
        __syncthreads();
        const float rm = fmaxf(fmaxf(p1[r], p1[256 + r]), fmaxf(p1[512 + r], p1[768 + r]));
        float se = 0.f, sen = 0.f, sb = 0.f;
        for (int j = q; j < tc; j += 4) {
          const int node = (j < t) ? (toff + j) : (coff + (j - t));
          const float nb = hbuf[node * D + r];
          float la, lb;
          if (j < t) { la = QA[j*D + r] + tA; lb = QB[j*D + r] + tB; }
          else { const int jc = j - t; la = PAb[jc*D + r] + cA; lb = PBb[jc*D + r] + cB; }
          const float ea = __expf(la - rm);
          se += ea; sen += ea * nb;
          sb += (lb / (1.f + fabsf(lb))) * nb;
        }
        __syncthreads();                 // all rm reads done before p1 reuse
        p1[tid] = se; p2[tid] = sen; p3[tid] = sb;
        __syncthreads();
        if (q == 0) {
          const float SE  = p1[r] + p1[256 + r] + p1[512 + r] + p1[768 + r];
          const float SEN = p2[r] + p2[256 + r] + p2[512 + r] + p2[768 + r];
          const float SB  = p3[r] + p3[256 + r] + p3[512 + r] + p3[768 + r];
          const float mv  = Wv[r] * (SEN / SE) + SB;
          m_lds[r] = mv;
          m_h[r]   = (_Float16)mv;
        }
      }
      __syncthreads();

      // phase 6: [Wu1;Wo1] @ m + v tail -> u (sigmoid) / o (relu): 32 tiles, one per wave
      {
        v8f acc = matvec_tile(S2h, m_h, wid);
        if ((lane & 15) == 0) {
          const int rbase = wid * 16 + (lane >> 4) * 8;
          #pragma unroll
          for (int i = 0; i < 8; ++i) {
            const int r = rbase + i;
            const float a = acc[i] + v_lds[1024 + r];   // Wu2@s+b_u or Wo2@s+b_o
            uo_lds[r] = (r < 256) ? (1.f / (1.f + __expf(-a))) : fmaxf(a, 0.f);
          }
        }
      }
      __syncthreads();

      // phase 7: new = u*o + (1-u)*m ; commit C[:,k]
      if (tid < 256) {
        const float u = uo_lds[tid], o = uo_lds[256 + tid], mm = m_lds[tid];
        const float nw = u * o + (1.f - u) * mm;
        new_h[tid] = (_Float16)nw;
        hbuf[(coff + k) * D + tid] = nw;
      }
      __syncthreads();

      // phase 8: PA[:,k] = WCA1@new ; PB[:,k] = WCB1@new  (32 tiles, one per wave)
      {
        v8f acc = matvec_tile(wca1, new_h, wid);     // rows 0-255 WCA1, 256-511 WCB1
        if ((lane & 15) == 0) {
          const int rbase = wid * 16 + (lane >> 4) * 8;
          #pragma unroll
          for (int i = 0; i < 8; ++i) {
            const int r = rbase + i;
            if (r < 256) PAb[k*D + r] = acc[i];
            else         PBb[k*D + (r - 256)] = acc[i];
          }
        }
      }
      __syncthreads();
    }
  }

  // epilogue: out is (D, N) row-major; hbuf is node-major
  for (int i = tid; i < D * NN; i += NTHREADS) {
    const int d = i / NN, n = i % NN;
    out[i] = hbuf[n * D + d];
  }
}

extern "C" void kernel_launch(void* const* d_in, const int* in_sizes, int n_in,
                              void* d_out, int out_size, void* d_ws, size_t ws_size,
                              hipStream_t stream) {
  const float* x    = (const float*)d_in[0];
  // d_in[1] = edge_index: unused by the reference
  const float* Wv   = (const float*)d_in[2];
  const float* W_TA = (const float*)d_in[3];
  const float* bTA  = (const float*)d_in[4];
  const float* W_CA = (const float*)d_in[5];
  const float* bCA  = (const float*)d_in[6];
  const float* W_TB = (const float*)d_in[7];
  const float* bTB  = (const float*)d_in[8];
  const float* W_CB = (const float*)d_in[9];
  const float* bCB  = (const float*)d_in[10];
  const float* W_u  = (const float*)d_in[11];
  const float* bu   = (const float*)d_in[12];
  const float* W_o  = (const float*)d_in[13];
  const float* bo   = (const float*)d_in[14];

  char* ws = (char*)d_ws;
  size_t off = 0;
  auto alloc = [&](size_t bytes) -> void* {
    void* p = ws + off;
    off += (bytes + 255) & ~(size_t)255;
    return p;
  };
  float*    hbuf = (float*)   alloc((size_t)NN * D * 4);
  _Float16* act  = (_Float16*)alloc((size_t)NN * D * 2);
  _Float16* wta1 = (_Float16*)alloc((size_t)D * D * 2);
  _Float16* wtb1 = (_Float16*)alloc((size_t)D * D * 2);
  _Float16* wc1h = (_Float16*)alloc((size_t)2 * D * D * 2);   // [WCA1;WCB1]
  _Float16* S1h  = (_Float16*)alloc((size_t)6 * D * D * 2);   // 1536x256
  _Float16* S2h  = (_Float16*)alloc((size_t)2 * D * D * 2);   // 512x256
  float*    QA   = (float*)   alloc((size_t)640 * D * 4);
  float*    QB   = (float*)   alloc((size_t)640 * D * 4);
  float*    PAb  = (float*)   alloc((size_t)640 * D * 4);
  float*    PBb  = (float*)   alloc((size_t)640 * D * 4);

  prep_kernel<<<128, 256, 0, stream>>>(x, W_TA, W_CA, W_TB, W_CB, W_u, W_o,
                                       hbuf, wta1, wtb1, wc1h, S1h, S2h);

  tree_main<<<1, NTHREADS, 0, stream>>>(hbuf, act, wta1, wtb1, wc1h,
                                        S1h, S2h, Wv,
                                        bTA, bCA, bTB, bCB, bu, bo,
                                        QA, QB, PAb, PBb, (float*)d_out);
}
